// AutoregressiveForecaster_25838523253160
// MI455X (gfx1250) — matrix operand order repaired and checked by reading, not compiled, not run on hardware
//
#include <hip/hip_runtime.h>

typedef __attribute__((ext_vector_type(16))) _Float16 v16h;
typedef __attribute__((ext_vector_type(8)))  _Float16 v8h;
typedef __attribute__((ext_vector_type(8)))  float    v8f;
typedef __attribute__((ext_vector_type(4)))  float    v4f;

#define HID            64
#define TLEN           24
#define STEPS          20
#define BATCHN         8192
#define WAVES          4
#define BLOCKT         (WAVES * 32)
#define ROWS_PER_WAVE  16
#define ROWS_PER_BLOCK (WAVES * ROWS_PER_WAVE)
#define XSTRIDE        48

// ---- dynamic LDS layout: hot per-wave buffers FIRST (16-bit DS imm offsets),
// ---- bulk weight fragments above 32 KB (per-matrix base VGPR + imm offsets).
#define OFF_HBUF   0u                            // per-wave 16x64 f16  (4 x 2 KB)
#define OFF_XSEQ   (OFF_HBUF + WAVES * 2048u)    // per-wave 16x48 f32  (4 x 3 KB)
#define OFF_MLP    (OFF_XSEQ + WAVES * 3072u)    // per-wave 16x32 f32  (4 x 2 KB)
#define OFF_B0     (OFF_MLP + WAVES * 2048u)     // 256 f32
#define OFF_B1     (OFF_B0 + 1024u)
#define OFF_W0C    (OFF_B1 + 1024u)
#define OFF_W2     (OFF_W0C + 1024u)
#define OFF_B1M    (OFF_W2 + 128u)               // ends < 32768
#define OFF_WHH0   32768u                        // 64x256 f16 B-frags (32 KB)
#define OFF_WIH1   (OFF_WHH0 + 32768u)
#define OFF_WHH1   (OFF_WIH1 + 32768u)
#define OFF_W1     (OFF_WHH1 + 32768u)           // 64x32 f16 (4 KB)
#define SMEM_BYTES (OFF_W1 + 4096u)              // 135168 B  (< 320 KB / WGP)

// ---- fast activations: native v_tanh_f32 / v_rcp_f32 where available ----
__device__ __forceinline__ float fast_tanh(float x) {
#if __has_builtin(__builtin_amdgcn_tanhf)
    return __builtin_amdgcn_tanhf(x);
#elif __has_builtin(__builtin_amdgcn_tanh_f32)
    return __builtin_amdgcn_tanh_f32(x);
#else
    float t = __expf(-2.0f * x);
    return (1.0f - t) * __builtin_amdgcn_rcpf(1.0f + t);
#endif
}
__device__ __forceinline__ float fast_sigm(float x) {
#if __has_builtin(__builtin_amdgcn_tanhf) || __has_builtin(__builtin_amdgcn_tanh_f32)
    return fmaf(0.5f, fast_tanh(0.5f * x), 0.5f);
#else
    return __builtin_amdgcn_rcpf(1.0f + __expf(-x));
#endif
}
// wave-private LDS ordering: HW keeps same-wave DS ops in order (ISA §7.3);
// this is only a compiler reordering barrier, no cross-wave sync needed.
__device__ __forceinline__ void wave_lds_fence() {
    __builtin_amdgcn_fence(__ATOMIC_ACQ_REL, "wavefront");
}
__device__ __forceinline__ v8f wmma16(v16h a, v16h b, v8f c) {
    return __builtin_amdgcn_wmma_f32_16x16x32_f16(
        false, a, false, b, (short)0, c, false, false);
}
__device__ __forceinline__ v16h load_afrag(const _Float16* hb, int col, int hi, int kt) {
    // A-layout (16-bit 16x32): lane row m=col; halves i<8 -> k=kt*32+hi*8+i,
    // halves i>=8 -> k=kt*32+16+hi*8+(i-8). Both 16B-contiguous in row-major hb.
    const v8h lo = *(const v8h*)(hb + col * HID + kt * 32 + hi * 8);
    const v8h hp = *(const v8h*)(hb + col * HID + kt * 32 + 16 + hi * 8);
    v16h a;
#pragma unroll
    for (int j = 0; j < 8; ++j) { a[j] = lo[j]; a[8 + j] = hp[j]; }
    return a;
}

__global__ __launch_bounds__(BLOCKT) void lstm_forecast_kernel(
    const float* __restrict__ x,
    const float* __restrict__ Wih0, const float* __restrict__ Whh0,
    const float* __restrict__ bih0, const float* __restrict__ bhh0,
    const float* __restrict__ Wih1, const float* __restrict__ Whh1,
    const float* __restrict__ bih1, const float* __restrict__ bhh1,
    const float* __restrict__ W1, const float* __restrict__ b1,
    const float* __restrict__ W2, const float* __restrict__ b2,
    const float* __restrict__ damping,
    float* __restrict__ out)
{
    extern __shared__ char smem[];
    const int tid  = threadIdx.x;
    const int lane = tid & 31;
    const int wave = tid >> 5;
    const int col  = lane & 15;   // N within tile / A row m
    const int hi   = lane >> 4;   // lane-half select

    _Float16* wz[3] = { (_Float16*)(smem + OFF_WHH0),
                        (_Float16*)(smem + OFF_WIH1),
                        (_Float16*)(smem + OFF_WHH1) };
    _Float16* w1z   = (_Float16*)(smem + OFF_W1);
    float* b0lds  = (float*)(smem + OFF_B0);
    float* b1lds  = (float*)(smem + OFF_B1);
    float* w0c    = (float*)(smem + OFF_W0C);
    float* w2lds  = (float*)(smem + OFF_W2);
    float* b1mlds = (float*)(smem + OFF_B1M);

    // ---------------- one-time prep: f32 -> f16 swizzled B fragments ----------------
    {
        const float* srcs[3] = { Whh0, Wih1, Whh1 };  // each [256][64], B(k,n)=W[n][k]
#pragma unroll
        for (int m = 0; m < 3; ++m) {
            const float* S = srcs[m];
            _Float16* D = wz[m];
            for (int e = tid; e < 16384; e += BLOCKT) {
                int fragid = e >> 9, r = e & 511;
                int l = r >> 4, i = r & 15;
                int nt = fragid >> 1, kt = fragid & 1;
                int k = (kt << 5) + ((i >> 3) << 4) + ((l >> 4) << 3) + (i & 7);   // B layout: k = (i/8)*16 + hi*8 + i%8
                int n = (nt << 4) + (l & 15);
                D[e] = (_Float16)S[n * HID + k];
            }
        }
        for (int e = tid; e < 2048; e += BLOCKT) {        // W1 [32][64]
            int fragid = e >> 9, r = e & 511;
            int l = r >> 4, i = r & 15;
            int nt = fragid >> 1, kt = fragid & 1;
            int k = (kt << 5) + ((i >> 3) << 4) + ((l >> 4) << 3) + (i & 7);
            int n = (nt << 4) + (l & 15);
            w1z[e] = (_Float16)W1[n * HID + k];
        }
        for (int j = tid; j < 256; j += BLOCKT) {
            b0lds[j] = bih0[j] + bhh0[j];
            b1lds[j] = bih1[j] + bhh1[j];
            w0c[j]   = Wih0[j];
        }
        for (int j = tid; j < 32; j += BLOCKT) {
            w2lds[j]  = W2[j];
            b1mlds[j] = b1[j];
        }
        // seed per-wave sliding windows from x [B][24]
        const int rb = blockIdx.x * ROWS_PER_BLOCK;
        for (int e = tid; e < ROWS_PER_BLOCK * TLEN; e += BLOCKT) {
            int row = e / TLEN, p = e - row * TLEN;
            float* xsW = (float*)(smem + OFF_XSEQ + (row >> 4) * 3072u);
            xsW[(row & 15) * XSTRIDE + p] = x[(rb + row) * TLEN + p];
        }
    }
    __syncthreads();   // the only workgroup barrier: weights become visible

    // ---------------- per-lane broadcast constants ----------------
    float b0v[16], b1v[16], wv[16];
#pragma unroll
    for (int j = 0; j < 16; ++j) {
        b0v[j] = b0lds[j * 16 + col];
        b1v[j] = b1lds[j * 16 + col];
        wv[j]  = w0c[j * 16 + col];
    }
    float b1mv[2], w2v[2];
#pragma unroll
    for (int j = 0; j < 2; ++j) {
        b1mv[j] = b1mlds[j * 16 + col];
        w2v[j]  = w2lds[j * 16 + col];
    }
    const float alpha = fast_sigm(damping[0]);
    const float oneMA = 1.0f - alpha;
    const float amix  = alpha * 0.5f;
    const float b2s   = b2[0];

    const v16h* B0  = (const v16h*)wz[0];   // 32 frags x 32 v16h each
    const v16h* BI1 = (const v16h*)wz[1];
    const v16h* BH1 = (const v16h*)wz[2];
    const v16h* BW1 = (const v16h*)w1z;
    _Float16* hb = (_Float16*)(smem + OFF_HBUF + wave * 2048u);
    float*    xs = (float*)(smem + OFF_XSEQ + wave * 3072u);
    float*    mb = (float*)(smem + OFF_MLP  + wave * 2048u);

    const int rowbase = blockIdx.x * ROWS_PER_BLOCK + wave * ROWS_PER_WAVE;
    float prev = 0.0f;

    v16h zh; v8f zf;
#pragma unroll
    for (int j = 0; j < 16; ++j) zh[j] = (_Float16)0.0f;
#pragma unroll
    for (int j = 0; j < 8; ++j) zf[j] = 0.0f;

    for (int s = 0; s < STEPS; ++s) {
        v16h a0[2], a1[2];
        a0[0] = zh; a0[1] = zh; a1[0] = zh; a1[1] = zh;
        v8f c0r[4], c1r[4];
#pragma unroll
        for (int nt = 0; nt < 4; ++nt) { c0r[nt] = zf; c1r[nt] = zf; }

        for (int t = 0; t < TLEN; ++t) {
            // x window values for this lane's 8 C-layout rows
            const float* xrow = xs + hi * 8 * XSTRIDE + s + t;
            float xv[8];
#pragma unroll
            for (int r = 0; r < 8; ++r) xv[r] = xrow[r * XSTRIDE];

            // ---- layer 0: gates = h0 @ Whh0^T (+ x*Wih0 + b) ----
#pragma unroll
            for (int nt = 0; nt < 4; ++nt) {
                v16h bf[8];
#pragma unroll
                for (int g = 0; g < 4; ++g)
#pragma unroll
                    for (int kt = 0; kt < 2; ++kt)
                        bf[g * 2 + kt] = B0[(((nt + g * 4) * 2) + kt) * 32 + lane];
                v8f gi = zf, gf = zf, gg = zf, go = zf;
                gi = wmma16(a0[0], bf[0], gi); gi = wmma16(a0[1], bf[1], gi);
                gf = wmma16(a0[0], bf[2], gf); gf = wmma16(a0[1], bf[3], gf);
                gg = wmma16(a0[0], bf[4], gg); gg = wmma16(a0[1], bf[5], gg);
                go = wmma16(a0[0], bf[6], go); go = wmma16(a0[1], bf[7], go);
#pragma unroll
                for (int r = 0; r < 8; ++r) {
                    float iv = gi[r] + b0v[nt]      + xv[r] * wv[nt];
                    float fv = gf[r] + b0v[nt +  4] + xv[r] * wv[nt +  4];
                    float gv = gg[r] + b0v[nt +  8] + xv[r] * wv[nt +  8];
                    float ov = go[r] + b0v[nt + 12] + xv[r] * wv[nt + 12];
                    float cc = fast_sigm(fv) * c0r[nt][r] + fast_sigm(iv) * fast_tanh(gv);
                    c0r[nt][r] = cc;
                    hb[(hi * 8 + r) * HID + nt * 16 + col] =
                        (_Float16)(fast_sigm(ov) * fast_tanh(cc));
                }
            }
            wave_lds_fence();
#pragma unroll
            for (int kt = 0; kt < 2; ++kt) a0[kt] = load_afrag(hb, col, hi, kt);
            wave_lds_fence();

            // ---- layer 1: gates = h0_new @ Wih1^T + h1 @ Whh1^T + b ----
#pragma unroll
            for (int nt = 0; nt < 4; ++nt) {
                v8f gi = zf, gf = zf, gg = zf, go = zf;
                {
                    v16h bf[8];
#pragma unroll
                    for (int g = 0; g < 4; ++g)
#pragma unroll
                        for (int kt = 0; kt < 2; ++kt)
                            bf[g * 2 + kt] = BI1[(((nt + g * 4) * 2) + kt) * 32 + lane];
                    gi = wmma16(a0[0], bf[0], gi); gi = wmma16(a0[1], bf[1], gi);
                    gf = wmma16(a0[0], bf[2], gf); gf = wmma16(a0[1], bf[3], gf);
                    gg = wmma16(a0[0], bf[4], gg); gg = wmma16(a0[1], bf[5], gg);
                    go = wmma16(a0[0], bf[6], go); go = wmma16(a0[1], bf[7], go);
                }
                {
                    v16h bf[8];
#pragma unroll
                    for (int g = 0; g < 4; ++g)
#pragma unroll
                        for (int kt = 0; kt < 2; ++kt)
                            bf[g * 2 + kt] = BH1[(((nt + g * 4) * 2) + kt) * 32 + lane];
                    gi = wmma16(a1[0], bf[0], gi); gi = wmma16(a1[1], bf[1], gi);
                    gf = wmma16(a1[0], bf[2], gf); gf = wmma16(a1[1], bf[3], gf);
                    gg = wmma16(a1[0], bf[4], gg); gg = wmma16(a1[1], bf[5], gg);
                    go = wmma16(a1[0], bf[6], go); go = wmma16(a1[1], bf[7], go);
                }
#pragma unroll
                for (int r = 0; r < 8; ++r) {
                    float iv = gi[r] + b1v[nt];
                    float fv = gf[r] + b1v[nt +  4];
                    float gv = gg[r] + b1v[nt +  8];
                    float ov = go[r] + b1v[nt + 12];
                    float cc = fast_sigm(fv) * c1r[nt][r] + fast_sigm(iv) * fast_tanh(gv);
                    c1r[nt][r] = cc;
                    hb[(hi * 8 + r) * HID + nt * 16 + col] =
                        (_Float16)(fast_sigm(ov) * fast_tanh(cc));
                }
            }
            wave_lds_fence();
#pragma unroll
            for (int kt = 0; kt < 2; ++kt) a1[kt] = load_afrag(hb, col, hi, kt);
            wave_lds_fence();
        } // t

        // ---- MLP head on h2[T-1] (a1 frags): relu(last@W1^T + b1) @ W2^T + b2 ----
#pragma unroll
        for (int nt2 = 0; nt2 < 2; ++nt2) {
            v8f acc = zf;
#pragma unroll
            for (int kt = 0; kt < 2; ++kt)
                acc = wmma16(a1[kt], BW1[(nt2 * 2 + kt) * 32 + lane], acc);
#pragma unroll
            for (int r = 0; r < 8; ++r) {
                float v0 = acc[r] + b1mv[nt2];
                v0 = fmaxf(v0, 0.0f) * w2v[nt2];
                mb[(hi * 8 + r) * 32 + nt2 * 16 + col] = v0;
            }
        }
        wave_lds_fence();
        if (lane < 16) {
            const v4f* mrow = (const v4f*)(mb + lane * 32);
            v4f acc4 = mrow[0];
#pragma unroll
            for (int q = 1; q < 8; ++q) acc4 += mrow[q];
            float p = b2s + acc4[0] + acc4[1] + acc4[2] + acc4[3];
            if (s > 0) p = p * oneMA + prev * amix;
            prev = p;
            xs[lane * XSTRIDE + TLEN + s] = p;                // append to window
            out[(rowbase + lane) * STEPS + s] = p;            // [B, steps]
        }
        wave_lds_fence();
    } // s
}

extern "C" void kernel_launch(void* const* d_in, const int* in_sizes, int n_in,
                              void* d_out, int out_size, void* d_ws, size_t ws_size,
                              hipStream_t stream) {
    (void)in_sizes; (void)n_in; (void)out_size; (void)d_ws; (void)ws_size;
    const float* x     = (const float*)d_in[0];
    const float* Wih0  = (const float*)d_in[1];
    const float* Whh0  = (const float*)d_in[2];
    const float* bih0  = (const float*)d_in[3];
    const float* bhh0  = (const float*)d_in[4];
    const float* Wih1  = (const float*)d_in[5];
    const float* Whh1  = (const float*)d_in[6];
    const float* bih1  = (const float*)d_in[7];
    const float* bhh1  = (const float*)d_in[8];
    const float* W1    = (const float*)d_in[9];
    const float* b1    = (const float*)d_in[10];
    const float* W2    = (const float*)d_in[11];
    const float* b2    = (const float*)d_in[12];
    const float* damp  = (const float*)d_in[13];
    float* out = (float*)d_out;

    // Opt in to >64KB dynamic LDS (gfx1250 WGP has 320 KB).
    static bool attr_set = false;
    if (!attr_set) {
        hipFuncSetAttribute((const void*)lstm_forecast_kernel,
                            hipFuncAttributeMaxDynamicSharedMemorySize, SMEM_BYTES);
        attr_set = true;
    }

    dim3 grid(BATCHN / ROWS_PER_BLOCK);   // 128 blocks
    dim3 block(BLOCKT);                   // 128 threads = 4 waves
    lstm_forecast_kernel<<<grid, block, SMEM_BYTES, stream>>>(
        x, Wih0, Whh0, bih0, bhh0, Wih1, Whh1, bih1, bhh1,
        W1, b1, W2, b2, damp, out);
}
